// ViTMoE_12197707120947
// MI455X (gfx1250) — matrix operand order repaired and checked by reading, not compiled, not run on hardware
//
#include <hip/hip_runtime.h>
#include <hip/hip_bf16.h>
#include <math.h>

typedef __attribute__((ext_vector_type(16))) _Float16 v16h;
typedef __attribute__((ext_vector_type(8)))  _Float16 v8h;
typedef __attribute__((ext_vector_type(8)))  float    v8f;

#define WAVES   8
#define IMGW    224
#define PLANE   50176   // 224*224
#define TTOK    1568    // 8*196
#define EDIM    384
#define NHEAD   6
#define DH      64
#define HIDD    2304
#define NEXP    6
#define NPATCH  196
#define NCLS    1000

enum { EP_BIAS = 1, EP_GELU = 2, EP_RES = 4, EP_CW = 8 };

__device__ __forceinline__ float gelu_erf(float x) {
  return 0.5f * x * (1.0f + erff(x * 0.70710678118654752f));
}

// A-fragment for v_wmma_f32_16x16x32_f16 (16x32 f16, MxK):
// lane l holds row m = l&15; halves 0..15 = K[koff..koff+15]
// with koff = (l>=16 ? 16 : 0).  `base` must already include m*K + k0.
__device__ __forceinline__ v16h load_a_frag(const _Float16* __restrict__ base, int lane) {
  const int koff = (lane >> 4) << 4;
  union { v16h v; v8h h[2]; } u;
  u.h[0] = *(const v8h*)(base + koff);
  u.h[1] = *(const v8h*)(base + koff + 8);
  return u.v;
}

// ---------------------------------------------------------------------------
// Generic WMMA GEMM:  out[n, m] = epilogue( sum_k X[n,k] * W[m,k] )
// W: [M,K] f16 row-major (K contiguous), X: [N,K] f16 row-major.
// One wave computes a 16(M) x 32(N) tile: one A fragment feeds two WMMAs.
// N must be a multiple of 32, K a multiple of 32.
// ---------------------------------------------------------------------------
__global__ void __launch_bounds__(256)
gemm_wt_f16(const _Float16* __restrict__ W, const _Float16* __restrict__ X,
            const float* __restrict__ bias, const float* __restrict__ res,
            const float* __restrict__ cw, int cwStride, int cwOff,
            float* __restrict__ outF, _Float16* __restrict__ outH,
            int M, int N, int K, int flags) {
  __shared__ v16h Bs[WAVES][32][2];
  const int wid  = threadIdx.x >> 5;
  const int lane = threadIdx.x & 31;
  const int tilesM = M >> 4;
  const int nTiles = tilesM * (N >> 5);
  int tile = blockIdx.x * WAVES + wid;
  const bool valid = tile < nTiles;
  if (!valid) tile = 0;
  const int m0 = (tile % tilesM) << 4;
  const int n0 = (tile / tilesM) << 5;

  const _Float16* Wp  = W + (size_t)(m0 + (lane & 15)) * K;
  const _Float16* Xp0 = X + (size_t)(n0 + (lane >> 1)) * K + ((lane & 1) << 4);
  const _Float16* Xp1 = Xp0 + (size_t)16 * K;

  v8f acc[2];
  acc[0] = (v8f){};
  acc[1] = (v8f){};
  for (int k0 = 0; k0 < K; k0 += 32) {
    Bs[wid][lane >> 1][lane & 1]        = *(const v16h*)(Xp0 + k0);
    Bs[wid][16 + (lane >> 1)][lane & 1] = *(const v16h*)(Xp1 + k0);
    if (k0 + 128 < K) {
      __builtin_prefetch((const void*)(Xp0 + k0 + 128), 0, 0);
      __builtin_prefetch((const void*)(Xp1 + k0 + 128), 0, 0);
    }
    __syncthreads();
    v16h a  = load_a_frag(Wp + k0, lane);
    v16h b0 = Bs[wid][lane & 15][lane >> 4];
    v16h b1 = Bs[wid][16 + (lane & 15)][lane >> 4];
    acc[0] = __builtin_amdgcn_wmma_f32_16x16x32_f16(false, a, false, b0, (short)0,
                                                    acc[0], false, false);
    acc[1] = __builtin_amdgcn_wmma_f32_16x16x32_f16(false, a, false, b1, (short)0,
                                                    acc[1], false, false);
    __syncthreads();
  }
  if (!valid) return;

  const int mB = m0 + ((lane >> 4) << 3);
#pragma unroll
  for (int sub = 0; sub < 2; ++sub) {
    const int nn = n0 + (sub << 4) + (lane & 15);
    const float cwv = (flags & EP_CW) ? cw[(size_t)nn * cwStride + cwOff] : 0.f;
#pragma unroll
    for (int j = 0; j < 8; ++j) {
      const int mm = mB + j;
      float v = acc[sub][j];
      if (flags & EP_BIAS) v += bias[mm];
      if (flags & EP_GELU) v = gelu_erf(v);
      const size_t oi = (size_t)nn * M + mm;
      if (flags & EP_CW) {
        outF[oi] += cwv * v;             // each (n,m) unique per launch
      } else {
        if (flags & EP_RES) v += res[oi];
        if (outF) outF[oi] = v;
        if (outH) outH[oi] = (_Float16)v;
      }
    }
  }
}

// ---------------------------------------------------------------------------
// 3x3 SAME conv as implicit GEMM (WMMA).  in: [B,Cin,224,224] f16,
// Wp packed: [Cout][9 taps][Cin] f16.  Epilogue: +bias, BN(eval), erf-GELU.
// One wave: 16 out-channels x 32 pixels (one row, 224 = 7*32).
// Uniform K loop for all waves (OOB taps stage zeros) -> aligned barriers.
// ---------------------------------------------------------------------------
__global__ void __launch_bounds__(256)
conv3x3_wmma(const _Float16* __restrict__ in, const _Float16* __restrict__ Wp,
             const float* __restrict__ cb, const float* __restrict__ bg,
             const float* __restrict__ bb, _Float16* __restrict__ out,
             int Cin, int Cout, int nTiles) {
  __shared__ v16h Bs[WAVES][32][2];
  const int wid  = threadIdx.x >> 5;
  const int lane = threadIdx.x & 31;
  int tile = blockIdx.x * WAVES + wid;
  const bool valid = tile < nTiles;
  if (!valid) tile = 0;
  const int pt = tile % 1568;            // 224 rows x 7 x-tiles (32 px each)
  int r = tile / 1568;
  const int nCoT = Cout >> 4;
  const int co0 = (r % nCoT) << 4;
  const int b   = r / nCoT;
  const int y   = pt / 7;
  const int x0  = (pt % 7) << 5;
  const int KW  = 9 * Cin;

  const _Float16* WpL = Wp + (size_t)(co0 + (lane & 15)) * KW;
  const _Float16* inB = in + (size_t)b * Cin * PLANE;
  _Float16* Bf = (_Float16*)&Bs[wid][0][0];   // [pixel 0..31][k 0..31]

  v8f acc[2];
  acc[0] = (v8f){};
  acc[1] = (v8f){};
  for (int tap = 0; tap < 9; ++tap) {
    const int ky = tap / 3, kx = tap % 3;
    const int yy = y + ky - 1;
    const int xb = x0 + kx - 1;
    const bool rowOK = (yy >= 0) && (yy < IMGW);
    const int ysafe = rowOK ? yy : 0;
    for (int ci0 = 0; ci0 < Cin; ci0 += 32) {
      // stage B tile: lane = channel row (32 rows), 32 pixels each
      const _Float16* src = inB + (size_t)(ci0 + lane) * PLANE + (size_t)ysafe * IMGW + xb;
#pragma unroll
      for (int p = 0; p < 32; ++p) {
        const int xx = xb + p;
        _Float16 val = (_Float16)0.f;
        if (rowOK && xx >= 0 && xx < IMGW) val = src[p];
        Bf[p * 32 + lane] = val;
      }
      __syncthreads();
      v16h a  = load_a_frag(WpL + tap * Cin + ci0, lane);
      v16h b0 = Bs[wid][lane & 15][lane >> 4];
      v16h b1 = Bs[wid][16 + (lane & 15)][lane >> 4];
      acc[0] = __builtin_amdgcn_wmma_f32_16x16x32_f16(false, a, false, b0, (short)0,
                                                      acc[0], false, false);
      acc[1] = __builtin_amdgcn_wmma_f32_16x16x32_f16(false, a, false, b1, (short)0,
                                                      acc[1], false, false);
      __syncthreads();
    }
  }
  if (!valid) return;

  const float rs = rsqrtf(1.0f + 1e-5f);
#pragma unroll
  for (int sub = 0; sub < 2; ++sub) {
    const int px = x0 + (sub << 4) + (lane & 15);
#pragma unroll
    for (int j = 0; j < 8; ++j) {
      const int co = co0 + ((lane >> 4) << 3) + j;
      float v = acc[sub][j] + cb[co];
      v = v * (bg[co] * rs) + bb[co];
      v = gelu_erf(v);
      out[((size_t)(b * Cout + co) * IMGW + y) * IMGW + px] = (_Float16)v;
    }
  }
}

// ---------------------------------------------------------------------------
// Patch-embed proj conv (16x16 stride 16) as WMMA GEMM, K = 384*256 = 98304.
// Wp: proj_w converted to f16 ([co][ci][py][px], already K-contiguous).
// One wave: 16 channels x 32 patches; B stage = aligned 32B v16h loads.
// tok[n, m] = acc + proj_b[m] + pos_embed[n%196, m]    (24*49 = 1176 tiles)
// ---------------------------------------------------------------------------
__global__ void __launch_bounds__(256)
proj_wmma(const _Float16* __restrict__ h3, const _Float16* __restrict__ Wp,
          const float* __restrict__ pb, const float* __restrict__ pos,
          float* __restrict__ tok) {
  __shared__ v16h Bs[WAVES][32][2];
  const int wid  = threadIdx.x >> 5;
  const int lane = threadIdx.x & 31;
  const int tile = blockIdx.x * WAVES + wid;
  const int m0 = (tile % 24) << 4;
  const int n0 = (tile / 24) << 5;
  const int K  = EDIM * 256;

  // two patches staged per lane
  const int pjA = n0 + (lane >> 1);
  const int pjB = pjA + 16;
  const int bA = pjA / NPATCH, ppA = pjA % NPATCH;
  const int bB = pjB / NPATCH, ppB = pjB % NPATCH;
  const int pyA = (ppA / 14) << 4, pxA = (ppA % 14) << 4;
  const int pyB = (ppB / 14) << 4, pxB = (ppB % 14) << 4;
  const _Float16* WpL = Wp + (size_t)(m0 + (lane & 15)) * K;

  v8f acc[2];
  acc[0] = (v8f){};
  acc[1] = (v8f){};
  for (int k0 = 0; k0 < K; k0 += 32) {
    const int k  = k0 + ((lane & 1) << 4);
    const int ci = k >> 8;
    const int py = (k & 255) >> 4;       // 16 px values contiguous in memory
    const _Float16* srcA =
        h3 + (((size_t)(bA * EDIM + ci)) * IMGW + (pyA + py)) * IMGW + pxA;
    const _Float16* srcB =
        h3 + (((size_t)(bB * EDIM + ci)) * IMGW + (pyB + py)) * IMGW + pxB;
    Bs[wid][lane >> 1][lane & 1]        = *(const v16h*)srcA;
    Bs[wid][16 + (lane >> 1)][lane & 1] = *(const v16h*)srcB;
    __syncthreads();
    v16h a  = load_a_frag(WpL + k0, lane);
    v16h b0 = Bs[wid][lane & 15][lane >> 4];
    v16h b1 = Bs[wid][16 + (lane & 15)][lane >> 4];
    acc[0] = __builtin_amdgcn_wmma_f32_16x16x32_f16(false, a, false, b0, (short)0,
                                                    acc[0], false, false);
    acc[1] = __builtin_amdgcn_wmma_f32_16x16x32_f16(false, a, false, b1, (short)0,
                                                    acc[1], false, false);
    __syncthreads();
  }

#pragma unroll
  for (int sub = 0; sub < 2; ++sub) {
    const int nn  = n0 + (sub << 4) + (lane & 15);
    const int ppn = nn % NPATCH;
#pragma unroll
    for (int j = 0; j < 8; ++j) {
      const int mm = m0 + ((lane >> 4) << 3) + j;
      tok[(size_t)nn * EDIM + mm] = acc[sub][j] + pb[mm] + pos[(size_t)ppn * EDIM + mm];
    }
  }
}

// ------------------------- small VALU kernels ------------------------------

// conv1: 3->192, 3x3 SAME, + BN + GELU, f32 in, f16 out.  K=27: VALU is fine.
__global__ void conv1_stem(const float* __restrict__ x, const float* __restrict__ w,
                           const float* __restrict__ cb, const float* __restrict__ bg,
                           const float* __restrict__ bb, _Float16* __restrict__ out,
                           int total) {
  const int idx = blockIdx.x * 256 + threadIdx.x;
  if (idx >= total) return;
  const int xx = idx % IMGW;
  int r = idx / IMGW;
  const int y  = r % IMGW; r /= IMGW;
  const int co = r % 192;
  const int b  = r / 192;
  float s = cb[co];
  for (int ci = 0; ci < 3; ++ci) {
    const float* xp = x + (size_t)(b * 3 + ci) * PLANE;
    const float* wp = w + (co * 3 + ci) * 9;
#pragma unroll
    for (int ky = 0; ky < 3; ++ky) {
      const int yy = y + ky - 1;
      if (yy < 0 || yy >= IMGW) continue;
#pragma unroll
      for (int kx = 0; kx < 3; ++kx) {
        const int x2 = xx + kx - 1;
        if (x2 < 0 || x2 >= IMGW) continue;
        s += xp[(size_t)yy * IMGW + x2] * wp[ky * 3 + kx];
      }
    }
  }
  s = s * (bg[co] * rsqrtf(1.0f + 1e-5f)) + bb[co];
  out[idx] = (_Float16)gelu_erf(s);
}

// LayerNorm over last dim E, one token per block (256 threads).
__global__ void ln_kernel(const float* __restrict__ in, const float* __restrict__ g,
                          const float* __restrict__ b, float* __restrict__ outF,
                          _Float16* __restrict__ outH, int E) {
  __shared__ float red[256];
  const int t = blockIdx.x, tid = threadIdx.x;
  const float* row = in + (size_t)t * E;
  float s = 0.f;
  for (int i = tid; i < E; i += 256) s += row[i];
  red[tid] = s; __syncthreads();
  for (int st = 128; st > 0; st >>= 1) { if (tid < st) red[tid] += red[tid + st]; __syncthreads(); }
  const float mean = red[0] / E; __syncthreads();
  float v = 0.f;
  for (int i = tid; i < E; i += 256) { const float d = row[i] - mean; v += d * d; }
  red[tid] = v; __syncthreads();
  for (int st = 128; st > 0; st >>= 1) { if (tid < st) red[tid] += red[tid + st]; __syncthreads(); }
  const float rstd = rsqrtf(red[0] / E + 1e-5f);
  for (int i = tid; i < E; i += 256) {
    const float o = (row[i] - mean) * rstd * g[i] + b[i];
    if (outF) outF[(size_t)t * E + i] = o;
    if (outH) outH[(size_t)t * E + i] = (_Float16)o;
  }
}

__global__ void attn_scores(const float* __restrict__ qkv, float* __restrict__ sc, int total) {
  const int idx = blockIdx.x * 256 + threadIdx.x;
  if (idx >= total) return;
  const int kj = idx % NPATCH;
  int r = idx / NPATCH;
  const int qi = r % NPATCH; r /= NPATCH;
  const int h = r % NHEAD;
  const int b = r / NHEAD;
  const float* q = qkv + (size_t)(b * NPATCH + qi) * 1152 + h * DH;
  const float* k = qkv + (size_t)(b * NPATCH + kj) * 1152 + EDIM + h * DH;
  float s = 0.f;
#pragma unroll 8
  for (int d = 0; d < DH; ++d) s += q[d] * k[d];
  sc[idx] = s * 0.125f;                  // 1/sqrt(64)
}

__global__ void attn_softmax(float* __restrict__ sc, int rows) {
  const int rIdx = blockIdx.x * 256 + threadIdx.x;
  if (rIdx >= rows) return;
  float* row = sc + (size_t)rIdx * NPATCH;
  float mx = -1e30f;
  for (int i = 0; i < NPATCH; ++i) mx = fmaxf(mx, row[i]);
  float s = 0.f;
  for (int i = 0; i < NPATCH; ++i) { const float e = expf(row[i] - mx); row[i] = e; s += e; }
  const float inv = 1.f / s;
  for (int i = 0; i < NPATCH; ++i) row[i] *= inv;
}

__global__ void attn_av(const float* __restrict__ qkv, const float* __restrict__ sc,
                        _Float16* __restrict__ aoh, int total) {
  const int idx = blockIdx.x * 256 + threadIdx.x;
  if (idx >= total) return;
  const int d = idx % DH;
  int r = idx / DH;
  const int qi = r % NPATCH; r /= NPATCH;
  const int h = r % NHEAD;
  const int b = r / NHEAD;
  const float* a = sc + (((size_t)(b * NHEAD + h)) * NPATCH + qi) * NPATCH;
  const float* v = qkv + (size_t)(b * NPATCH) * 1152 + 2 * EDIM + h * DH + d;
  float s = 0.f;
  for (int kj = 0; kj < NPATCH; ++kj) s += a[kj] * v[(size_t)kj * 1152];
  aoh[(size_t)(b * NPATCH + qi) * EDIM + h * DH + d] = (_Float16)s;
}

// router softmax + top-2 -> combine weights cw[t, e]
__global__ void router_topk(const float* __restrict__ tok, const float* __restrict__ rw,
                            const float* __restrict__ rb, float* __restrict__ cw) {
  const int t = blockIdx.x * 256 + threadIdx.x;
  if (t >= TTOK) return;
  const float* x = tok + (size_t)t * EDIM;
  float lg[NEXP];
  for (int e = 0; e < NEXP; ++e) {
    const float* w = rw + e * EDIM;
    float s = rb[e];
    for (int i = 0; i < EDIM; ++i) s += x[i] * w[i];
    lg[e] = s;
  }
  float mx = lg[0];
  for (int e = 1; e < NEXP; ++e) mx = fmaxf(mx, lg[e]);
  float p[NEXP], s = 0.f;
  for (int e = 0; e < NEXP; ++e) { p[e] = expf(lg[e] - mx); s += p[e]; }
  const float inv = 1.f / s;
  for (int e = 0; e < NEXP; ++e) p[e] *= inv;
  int i1 = 0;
  for (int e = 1; e < NEXP; ++e) if (p[e] > p[i1]) i1 = e;
  int i2 = (i1 == 0) ? 1 : 0;
  for (int e = 0; e < NEXP; ++e) if (e != i1 && p[e] > p[i2]) i2 = e;
  for (int e = 0; e < NEXP; ++e)
    cw[(size_t)t * NEXP + e] = (e == i1 || e == i2) ? p[e] : 0.f;
}

__global__ void pool_kernel(const float* __restrict__ ln, float* __restrict__ pooled) {
  const int idx = blockIdx.x * 256 + threadIdx.x;
  if (idx >= 8 * EDIM) return;
  const int d = idx % EDIM, b = idx / EDIM;
  float s = 0.f;
  for (int t = 0; t < NPATCH; ++t) s += ln[((size_t)(b * NPATCH + t)) * EDIM + d];
  pooled[idx] = s * (1.0f / NPATCH);
}

__global__ void head_kernel(const float* __restrict__ pooled, const float* __restrict__ hw,
                            const float* __restrict__ hb, float* __restrict__ out) {
  const int idx = blockIdx.x * 256 + threadIdx.x;
  if (idx >= 8 * NCLS) return;
  const int c = idx % NCLS, b = idx / NCLS;
  const float* p = pooled + (size_t)b * EDIM;
  const float* w = hw + (size_t)c * EDIM;
  float s = hb[c];
  for (int i = 0; i < EDIM; ++i) s += p[i] * w[i];
  out[idx] = s;
}

// pack conv weight [Cout][Cin][3][3] f32 -> [Cout][tap][Cin] f16
__global__ void pack_conv_w(const float* __restrict__ src, _Float16* __restrict__ dst,
                            int Cin, int total) {
  const int idx = blockIdx.x * 256 + threadIdx.x;
  if (idx >= total) return;
  const int ci = idx % Cin;
  int r = idx / Cin;
  const int tap = r % 9;
  const int co = r / 9;
  dst[idx] = (_Float16)src[(size_t)(co * Cin + ci) * 9 + tap];
}

__global__ void cvt_f16(const float* __restrict__ src, _Float16* __restrict__ dst, int total) {
  const int idx = blockIdx.x * 256 + threadIdx.x;
  if (idx < total) dst[idx] = (_Float16)src[idx];
}

__global__ void zero_f32(float* __restrict__ p, int total) {
  const int idx = blockIdx.x * 256 + threadIdx.x;
  if (idx < total) p[idx] = 0.f;
}

// ---------------------------------------------------------------------------
extern "C" void kernel_launch(void* const* d_in, const int* in_sizes, int n_in,
                              void* d_out, int out_size, void* d_ws, size_t ws_size,
                              hipStream_t stream) {
  (void)in_sizes; (void)n_in; (void)out_size; (void)ws_size;
  const float* x    = (const float*)d_in[0];
  const float* c1w  = (const float*)d_in[1];  const float* c1b = (const float*)d_in[2];
  const float* bn1g = (const float*)d_in[3];  const float* bn1b = (const float*)d_in[4];
  const float* c2w  = (const float*)d_in[5];  const float* c2b = (const float*)d_in[6];
  const float* bn2g = (const float*)d_in[7];  const float* bn2b = (const float*)d_in[8];
  const float* c3w  = (const float*)d_in[9];  const float* c3b = (const float*)d_in[10];
  const float* bn3g = (const float*)d_in[11]; const float* bn3b = (const float*)d_in[12];
  const float* pw   = (const float*)d_in[13]; const float* pbias = (const float*)d_in[14];
  const float* pos  = (const float*)d_in[15];
  const float* ln1g = (const float*)d_in[16]; const float* ln1b = (const float*)d_in[17];
  const float* ipw  = (const float*)d_in[18]; const float* ipb = (const float*)d_in[19];
  const float* opw  = (const float*)d_in[20]; const float* opb = (const float*)d_in[21];
  const float* rw   = (const float*)d_in[22]; const float* rb = (const float*)d_in[23];
  const float* ew1  = (const float*)d_in[24]; const float* eb1 = (const float*)d_in[25];
  const float* ew2  = (const float*)d_in[26]; const float* eb2 = (const float*)d_in[27];
  const float* ln2g = (const float*)d_in[28]; const float* ln2b = (const float*)d_in[29];
  const float* hw   = (const float*)d_in[30]; const float* hb = (const float*)d_in[31];

  char* base = (char*)d_ws;
  size_t off = 0;
  auto alloc = [&](size_t bytes) -> void* {
    void* p = base + off;
    off = (off + bytes + 255) & ~(size_t)255;
    return p;
  };

  const size_t actC  = (size_t)8 * 192 * PLANE;   // 77,070,336
  const size_t actE  = (size_t)8 * EDIM * PLANE;  // 154,140,672
  _Float16* h1   = (_Float16*)alloc(actC * 2);
  _Float16* h2   = (_Float16*)alloc(actC * 2);
  _Float16* h3   = (_Float16*)alloc(actE * 2);
  _Float16* w2p  = (_Float16*)alloc((size_t)192 * 9 * 192 * 2);
  _Float16* w3p  = (_Float16*)alloc((size_t)384 * 9 * 192 * 2);
  _Float16* wpp  = (_Float16*)alloc((size_t)384 * 98304 * 2);
  _Float16* wip  = (_Float16*)alloc((size_t)1152 * EDIM * 2);
  _Float16* wop  = (_Float16*)alloc((size_t)EDIM * EDIM * 2);
  _Float16* we1  = (_Float16*)alloc((size_t)NEXP * HIDD * EDIM * 2);
  _Float16* we2  = (_Float16*)alloc((size_t)NEXP * EDIM * HIDD * 2);
  float*    tok0 = (float*)alloc((size_t)TTOK * EDIM * 4);
  _Float16* nrmh = (_Float16*)alloc((size_t)TTOK * EDIM * 2);
  float*    qkv  = (float*)alloc((size_t)TTOK * 1152 * 4);
  float*    sc   = (float*)alloc((size_t)8 * NHEAD * NPATCH * NPATCH * 4);
  _Float16* aoh  = (_Float16*)alloc((size_t)TTOK * EDIM * 2);
  float*    tok1 = (float*)alloc((size_t)TTOK * EDIM * 4);
  _Float16* tokh = (_Float16*)alloc((size_t)TTOK * EDIM * 2);
  float*    cw   = (float*)alloc((size_t)TTOK * NEXP * 4);
  _Float16* eh   = (_Float16*)alloc((size_t)TTOK * HIDD * 2);
  float*    moe  = (float*)alloc((size_t)TTOK * EDIM * 4);
  float*    lnout= (float*)alloc((size_t)TTOK * EDIM * 4);
  float*    pooled = (float*)alloc((size_t)8 * EDIM * 4);
  float*    out  = (float*)d_out;

  auto blk = [](long n) { return dim3((unsigned)((n + 255) / 256)); };

  // --- weight pack / convert ---
  {
    int t2 = 192 * 9 * 192;
    pack_conv_w<<<blk(t2), 256, 0, stream>>>(c2w, w2p, 192, t2);
    int t3 = 384 * 9 * 192;
    pack_conv_w<<<blk(t3), 256, 0, stream>>>(c3w, w3p, 192, t3);
    cvt_f16<<<blk(384 * 98304), 256, 0, stream>>>(pw, wpp, 384 * 98304);
    cvt_f16<<<blk(1152 * EDIM), 256, 0, stream>>>(ipw, wip, 1152 * EDIM);
    cvt_f16<<<blk(EDIM * EDIM), 256, 0, stream>>>(opw, wop, EDIM * EDIM);
    cvt_f16<<<blk(NEXP * HIDD * EDIM), 256, 0, stream>>>(ew1, we1, NEXP * HIDD * EDIM);
    cvt_f16<<<blk(NEXP * EDIM * HIDD), 256, 0, stream>>>(ew2, we2, NEXP * EDIM * HIDD);
  }

  // --- conv stem ---
  conv1_stem<<<blk((long)actC), 256, 0, stream>>>(x, c1w, c1b, bn1g, bn1b, h1, (int)actC);
  {
    int tiles2 = 8 * (192 / 16) * 1568;   // 150,528
    conv3x3_wmma<<<dim3((tiles2 + WAVES - 1) / WAVES), 256, 0, stream>>>(
        h1, w2p, c2b, bn2g, bn2b, h2, 192, 192, tiles2);
    int tiles3 = 8 * (384 / 16) * 1568;   // 301,056
    conv3x3_wmma<<<dim3((tiles3 + WAVES - 1) / WAVES), 256, 0, stream>>>(
        h2, w3p, c3b, bn3g, bn3b, h3, 192, 384, tiles3);
  }
  // --- patch embed + pos ---
  proj_wmma<<<dim3((24 * 49) / WAVES + 1 - 1), 256, 0, stream>>>(h3, wpp, pbias, pos, tok0);

  // --- attention block ---
  ln_kernel<<<dim3(TTOK), 256, 0, stream>>>(tok0, ln1g, ln1b, nullptr, nrmh, EDIM);
  {
    int tiles = (1152 / 16) * (TTOK / 32);
    gemm_wt_f16<<<dim3((tiles + WAVES - 1) / WAVES), 256, 0, stream>>>(
        wip, nrmh, ipb, nullptr, nullptr, 0, 0, qkv, nullptr, 1152, TTOK, EDIM, EP_BIAS);
  }
  {
    int tS = 8 * NHEAD * NPATCH * NPATCH;
    attn_scores<<<blk(tS), 256, 0, stream>>>(qkv, sc, tS);
    attn_softmax<<<blk(8 * NHEAD * NPATCH), 256, 0, stream>>>(sc, 8 * NHEAD * NPATCH);
    int tA = 8 * NHEAD * NPATCH * DH;
    attn_av<<<blk(tA), 256, 0, stream>>>(qkv, sc, aoh, tA);
  }
  {
    int tiles = (EDIM / 16) * (TTOK / 32);
    gemm_wt_f16<<<dim3((tiles + WAVES - 1) / WAVES), 256, 0, stream>>>(
        wop, aoh, opb, tok0, nullptr, 0, 0, tok1, tokh, EDIM, TTOK, EDIM,
        EP_BIAS | EP_RES);
  }

  // --- MoE ---
  router_topk<<<blk(TTOK), 256, 0, stream>>>(tok1, rw, rb, cw);
  zero_f32<<<blk(TTOK * EDIM), 256, 0, stream>>>(moe, TTOK * EDIM);
  for (int e = 0; e < NEXP; ++e) {
    int tiles1 = (HIDD / 16) * (TTOK / 32);
    gemm_wt_f16<<<dim3((tiles1 + WAVES - 1) / WAVES), 256, 0, stream>>>(
        we1 + (size_t)e * HIDD * EDIM, tokh, eb1 + (size_t)e * HIDD, nullptr,
        nullptr, 0, 0, nullptr, eh, HIDD, TTOK, EDIM, EP_BIAS | EP_GELU);
    int tiles2 = (EDIM / 16) * (TTOK / 32);
    gemm_wt_f16<<<dim3((tiles2 + WAVES - 1) / WAVES), 256, 0, stream>>>(
        we2 + (size_t)e * EDIM * HIDD, eh, eb2 + (size_t)e * EDIM, nullptr,
        cw, NEXP, e, moe, nullptr, EDIM, TTOK, HIDD, EP_BIAS | EP_CW);
  }

  // --- final norm, pool, head ---
  ln_kernel<<<dim3(TTOK), 256, 0, stream>>>(moe, ln2g, ln2b, lnout, nullptr, EDIM);
  pool_kernel<<<blk(8 * EDIM), 256, 0, stream>>>(lnout, pooled);
  head_kernel<<<blk(8 * NCLS), 256, 0, stream>>>(pooled, hw, hb, out);
}
